// GraphEmbedder_57543971832471
// MI455X (gfx1250) — matrix operand order repaired and checked by reading, not compiled, not run on hardware
//
#include <hip/hip_runtime.h>
#include <hip/hip_bf16.h>

#define BB 8
#define NN 1024
#define EE 10
#define EP 16            // E padded to 16 for WMMA N-dimension
#define TSTRIDE (NN + 8) // padded LDS row stride (halves): 2064B = 516 dwords,
                         // 516 % 64 == 4 -> conflict-free b128 reads

typedef __attribute__((ext_vector_type(16))) _Float16 v16h;
typedef __attribute__((ext_vector_type(8)))  float    v8f;
typedef __attribute__((ext_vector_type(4)))  _Float16 h4;

// ---------------------------------------------------------------------------
// 1) Convert adjacency f32 -> f16 (16 MiB, stays hot in 192MB L2 for 5 iters)
// ---------------------------------------------------------------------------
__global__ void cvt_adj_kernel(const float* __restrict__ adj,
                               _Float16* __restrict__ adj_h) {
    size_t i = ((size_t)blockIdx.x * blockDim.x + threadIdx.x) * 4;
    float4 v = *(const float4*)(adj + i);
    h4 h;
    h[0] = (_Float16)v.x; h[1] = (_Float16)v.y;
    h[2] = (_Float16)v.z; h[3] = (_Float16)v.w;
    *(h4*)(adj_h + i) = h;
}

// ---------------------------------------------------------------------------
// 2) Column sums of weights: sum_i max(w,0) and sum_i min(w,0).
//    Exactly replaces the [B,N,N,E] relu tensor:
//    sum_i relu(w*c) = c>=0 ? c*sumpos : c*sumneg.
// ---------------------------------------------------------------------------
__global__ void colsum_kernel(const float* __restrict__ w,
                              float* __restrict__ cp, float* __restrict__ cn) {
    int t = blockIdx.x * blockDim.x + threadIdx.x;   // (b,j) flat, B*N total
    int b = t >> 10;
    int j = t & (NN - 1);
    const float* col = w + (size_t)b * NN * NN + j;
    float sp = 0.f, sn = 0.f;
    for (int i = 0; i < NN; ++i) {
        float x = col[(size_t)i * NN];
        sp += fmaxf(x, 0.f);
        sn += fminf(x, 0.f);
    }
    cp[t] = sp; cn[t] = sn;
}

// ---------------------------------------------------------------------------
// 3) base[b,j,f] = feat*w_sel[f] + sumpos*alpha[f] + sumneg*beta[f]
//    emb0 = relu(base)  (this IS iteration 1 of the reference, since emb=0).
//    f16 embedding buffer is written TRANSPOSED: [B][EP][N].
// ---------------------------------------------------------------------------
__global__ void base_init_kernel(const float* __restrict__ feat,
                                 const float* __restrict__ cp,
                                 const float* __restrict__ cn,
                                 const float* __restrict__ w_sel,
                                 const float* __restrict__ w_nb,
                                 const float* __restrict__ w_ew,
                                 float* __restrict__ basebuf,
                                 float* __restrict__ emb_f32,
                                 _Float16* __restrict__ emb_hT) {
    int t = blockIdx.x * blockDim.x + threadIdx.x;   // B*N
    int b = t >> 10;
    int j = t & (NN - 1);
    float f  = feat[t];
    float sp = cp[t], sn = cn[t];
    #pragma unroll
    for (int ff = 0; ff < EP; ++ff) {
        float v = 0.f;
        if (ff < EE) {
            float ap = 0.f, an = 0.f;
            #pragma unroll
            for (int e = 0; e < EE; ++e) {
                float c  = w_ew[e];
                float wn = w_nb[ff * EE + e];
                if (c >= 0.f) ap += wn * c; else an += wn * c;
            }
            v = f * w_sel[ff] + sp * ap + sn * an;
        }
        size_t o = (size_t)t * EP + ff;
        basebuf[o] = v;
        float r = fmaxf(v, 0.f);
        emb_f32[o] = r;
        emb_hT[((size_t)b * EP + ff) * NN + j] = (_Float16)r;
    }
}

// ---------------------------------------------------------------------------
// 4) One propagation iteration:
//    emb_out = relu(base + (A @ emb_in) @ Wp^T)
//    WMMA f16 16x16x32; per-wave 16-row M tile, N = E padded to 16, K = 1024.
//    emb_in is [B][EP][N] f16; staged transposed into padded LDS rows so each
//    B fragment is 2x ds_load_b128 (lane's 16 K-values: two contiguous runs of 8).
// ---------------------------------------------------------------------------
__global__ __launch_bounds__(64)
void iter_kernel(const _Float16* __restrict__ adj_h,
                 const _Float16* __restrict__ emb_inT,
                 const float* __restrict__ basebuf,
                 const float* __restrict__ w_nbpriors,
                 float* __restrict__ emb_out_f32,
                 _Float16* __restrict__ emb_outT) {
    __shared__ _Float16 lds_embT[EP * TSTRIDE]; // ~32.25 KB transposed operand
    __shared__ float    lds_p[2 * 16 * 16];     // per-wave accumulator tiles

    const int b        = blockIdx.y;
    const int rowBlock = blockIdx.x * 32;       // 2 waves * 16 rows
    const int tid      = threadIdx.x;

    // cooperative coalesced stage: global [EP][NN] row -> LDS padded row
    {
        const uint4* src = (const uint4*)(emb_inT + (size_t)b * EP * NN);
        #pragma unroll
        for (int e = 0; e < EP; ++e) {
            uint4* drow = (uint4*)(lds_embT + e * TSTRIDE);
            const uint4* srow = src + e * (NN / 8);
            for (int i = tid; i < NN / 8; i += 64) drow[i] = srow[i];
        }
    }
    __syncthreads();

    const int wave = tid >> 5;
    const int lane = tid & 31;
    const int half = lane >> 4;                 // K-half select
    const int lm   = lane & 15;
    const int row  = rowBlock + wave * 16 + lm; // A row held by this lane
    const _Float16* arow = adj_h + ((size_t)(b * NN + row)) * NN;
    const _Float16* brow = lds_embT + lm * TSTRIDE + 8 * half;

    v8f acc = {};
    #pragma unroll 2
    for (int ks = 0; ks < NN; ks += 32) {
        // A fragment: halves [ks+8*half, +8) and [ks+16+8*half, +8) -> 2x b128
        union { v16h h; uint4 q[2]; } A;
        A.q[0] = *(const uint4*)(arow + ks + 8 * half);
        A.q[1] = *(const uint4*)(arow + ks + 16 + 8 * half);
        // B fragment: column lm, halves [ks+8*half, +8) and [ks+16+8*half, +8) -> 2x ds_load_b128
        union { v16h h; uint4 q[2]; } Bv;
        Bv.q[0] = *(const uint4*)(brow + ks);
        Bv.q[1] = *(const uint4*)(brow + ks + 16);
        acc = __builtin_amdgcn_wmma_f32_16x16x32_f16(
            false, A.h, false, Bv.h, (short)0, acc, false, false);
    }

    // Bounce 16x16 f32 accumulator through LDS (D layout: VGPR r -> M=r+8*half, N=lm)
    float* pw = lds_p + wave * 256;
    #pragma unroll
    for (int r = 0; r < 8; ++r) pw[(r + 8 * half) * 16 + lm] = acc[r];
    __syncthreads();

    // Epilogue: out[i,f] = relu(base + sum_e Wp[f,e] * P[i,e])
    for (int idx = tid; idx < 32 * 16; idx += 64) {
        int il = idx >> 4, f = idx & 15;
        float p = 0.f;
        if (f < EE) {
            #pragma unroll
            for (int e = 0; e < EE; ++e)
                p += w_nbpriors[f * EE + e] * lds_p[il * 16 + e];
        }
        int r = rowBlock + il;
        size_t o = ((size_t)b * NN + r) * EP + f;
        float val = fmaxf(basebuf[o] + p, 0.f);
        emb_out_f32[o] = val;
        emb_outT[((size_t)b * EP + f) * NN + r] = (_Float16)val;
    }
}

// ---------------------------------------------------------------------------
// 5) S[b] = sum_{n,e} emb[b,n,e] * h[e], h[e] = sum_f wr[f]*Wall[f,e]
// ---------------------------------------------------------------------------
__global__ void reduce_s_kernel(const float* __restrict__ emb_f32,
                                const float* __restrict__ wr,
                                const float* __restrict__ wall,
                                float* __restrict__ Sout) {
    __shared__ float red[256];
    int b = blockIdx.x, tid = threadIdx.x;
    float h[EE];
    #pragma unroll
    for (int e = 0; e < EE; ++e) {
        float s = 0.f;
        #pragma unroll
        for (int f = 0; f < EE; ++f) s += wr[f] * wall[f * EE + e];
        h[e] = s;
    }
    float acc = 0.f;
    for (int n = tid; n < NN; n += 256) {
        const float* em = emb_f32 + ((size_t)b * NN + n) * EP;
        #pragma unroll
        for (int e = 0; e < EE; ++e) acc += em[e] * h[e];
    }
    red[tid] = acc; __syncthreads();
    for (int s = 128; s > 0; s >>= 1) {
        if (tid < s) red[tid] += red[tid + s];
        __syncthreads();
    }
    if (tid == 0) Sout[b] = red[0];
}

// ---------------------------------------------------------------------------
// 6) q[b,n] = S[b] + sum_e emb*g[e], g[e] = sum_f wr[E+f]*Wact[f,e];
//    also emit embeddings (unpadded) into d_out.
// ---------------------------------------------------------------------------
__global__ void qout_kernel(const float* __restrict__ emb_f32,
                            const float* __restrict__ wr,
                            const float* __restrict__ wact,
                            const float* __restrict__ S,
                            float* __restrict__ out) {
    int t = blockIdx.x * blockDim.x + threadIdx.x;   // B*N
    int b = t >> 10;
    float g[EE];
    #pragma unroll
    for (int e = 0; e < EE; ++e) {
        float s = 0.f;
        #pragma unroll
        for (int f = 0; f < EE; ++f) s += wr[EE + f] * wact[f * EE + e];
        g[e] = s;
    }
    const float* em = emb_f32 + (size_t)t * EP;
    float* eo = out + (size_t)BB * NN + (size_t)t * EE;
    float q = S[b];
    #pragma unroll
    for (int e = 0; e < EE; ++e) {
        float v = em[e];
        q += v * g[e];
        eo[e] = v;
    }
    out[t] = q;
}

// ---------------------------------------------------------------------------
extern "C" void kernel_launch(void* const* d_in, const int* in_sizes, int n_in,
                              void* d_out, int out_size, void* d_ws, size_t ws_size,
                              hipStream_t stream) {
    (void)in_sizes; (void)n_in; (void)out_size; (void)ws_size;
    const float* features  = (const float*)d_in[0];
    const float* weights   = (const float*)d_in[1];
    const float* adjacency = (const float*)d_in[2];
    const float* w_sel     = (const float*)d_in[3];
    const float* w_pr      = (const float*)d_in[4];
    const float* w_nb      = (const float*)d_in[5];
    const float* w_ew      = (const float*)d_in[6];
    const float* w_qr      = (const float*)d_in[7];
    const float* w_all     = (const float*)d_in[8];
    const float* w_act     = (const float*)d_in[9];

    char* ws = (char*)d_ws;
    size_t off = 0;
    _Float16* adj_h  = (_Float16*)(ws + off); off += (size_t)BB * NN * NN * sizeof(_Float16); // 16 MiB
    float* basebuf   = (float*)(ws + off);    off += (size_t)BB * NN * EP * sizeof(float);     // 512 KiB
    float* emb_f32   = (float*)(ws + off);    off += (size_t)BB * NN * EP * sizeof(float);     // 512 KiB
    _Float16* embhT0 = (_Float16*)(ws + off); off += (size_t)BB * EP * NN * sizeof(_Float16);  // 256 KiB
    _Float16* embhT1 = (_Float16*)(ws + off); off += (size_t)BB * EP * NN * sizeof(_Float16);  // 256 KiB
    float* colp      = (float*)(ws + off);    off += (size_t)BB * NN * sizeof(float);
    float* coln      = (float*)(ws + off);    off += (size_t)BB * NN * sizeof(float);
    float* Sbuf      = (float*)(ws + off);    off += 256;

    cvt_adj_kernel<<<(BB * NN * NN / 4) / 256, 256, 0, stream>>>(adjacency, adj_h);
    colsum_kernel<<<BB * NN / 256, 256, 0, stream>>>(weights, colp, coln);
    base_init_kernel<<<BB * NN / 256, 256, 0, stream>>>(
        features, colp, coln, w_sel, w_nb, w_ew, basebuf, emb_f32, embhT0);

    dim3 g(NN / 32, BB);
    // 4 remaining iterations (iteration 1 == relu(base), done in init)
    iter_kernel<<<g, 64, 0, stream>>>(adj_h, embhT0, basebuf, w_pr, emb_f32, embhT1);
    iter_kernel<<<g, 64, 0, stream>>>(adj_h, embhT1, basebuf, w_pr, emb_f32, embhT0);
    iter_kernel<<<g, 64, 0, stream>>>(adj_h, embhT0, basebuf, w_pr, emb_f32, embhT1);
    iter_kernel<<<g, 64, 0, stream>>>(adj_h, embhT1, basebuf, w_pr, emb_f32, embhT0);

    reduce_s_kernel<<<BB, 256, 0, stream>>>(emb_f32, w_qr, w_all, Sbuf);
    qout_kernel<<<BB * NN / 256, 256, 0, stream>>>(emb_f32, w_qr, w_act, Sbuf, (float*)d_out);
}
